// LocalSpatialAttention_59433757442673
// MI455X (gfx1250) — matrix operand order repaired and checked by reading, not compiled, not run on hardware
//
#include <hip/hip_runtime.h>
#include <hip/hip_bf16.h>
#include <math.h>

typedef __attribute__((ext_vector_type(16))) _Float16 v16h;
typedef __attribute__((ext_vector_type(4)))  _Float16 v4h;
typedef __attribute__((ext_vector_type(8)))  float    v8f;

namespace {
constexpr int B_ = 8, Y_ = 128, X_ = 128, C_ = 256, H_ = 8, F_ = 32, S_ = 25;
constexpr int NROW = B_ * Y_ * X_;            // 131072 pixels
constexpr float SCALE = 0.17677669529663687f; // 1/sqrt(F)
constexpr int PADH = 24;  // per-lane fragment stride in LDS (16 data + 8 pad halves)
}

__device__ __forceinline__ int clampi(int v, int lo, int hi) {
  return v < lo ? lo : (v > hi ? hi : v);
}
__device__ __forceinline__ int mini(int a, int b) { return a < b ? a : b; }

// ---- weight f32 -> f16, pre-swizzled into WMMA B-fragment order ----
// d[((tile*8 + chunk)*32 + lane)*16 + e] = (f16) s[(chunk*32 + (e&7) + ((e&8)<<1) + 8*hs)*256 + tile*16 + (lane&15)]
__global__ __launch_bounds__(256) void k_cvt_swz(const float* __restrict__ s,
                                                 _Float16* __restrict__ d) {
  int i = blockIdx.x * 256 + threadIdx.x;   // 0..65535
  int e    = i & 15;
  int lane = (i >> 4) & 31;
  int chunk = (i >> 9) & 7;
  int tile  = i >> 12;
  int hs = lane >> 4;
  int n  = (tile << 4) + (lane & 15);
  int k  = (chunk << 5) + (e & 7) + ((e & 8) << 1) + (hs << 3);
  d[i] = (_Float16)s[k * C_ + n];
}

// ---- stage a 16-row f32 stripe into LDS as f16 in A-fragment order ----
// A layout (16-bit, 16x32): lane = m + 16*hs; element e <-> k = (e&7) + ((e&8)<<1) + hs*8
__device__ __forceinline__ void stage_a_swz(const float* __restrict__ src,
                                            _Float16* __restrict__ aswz, int tid) {
  const float4* s4 = (const float4*)src;
  for (int g = tid; g < 16 * C_ / 4; g += 256) {   // 1024 float4 groups
    int m  = g >> 6;        // row 0..15
    int k4 = g & 63;        // k/4
    float4 v = s4[g];
    int chunk = k4 >> 3;
    int hs    = (k4 >> 1) & 1;
    int e0    = ((k4 & 1) << 2) | (((k4 >> 2) & 1) << 3);
    int off   = ((chunk << 5) + m + (hs << 4)) * PADH + e0;
    v4h hv = {(_Float16)v.x, (_Float16)v.y, (_Float16)v.z, (_Float16)v.w};
    *(v4h*)(aswz + off) = hv;
  }
}

// ---------------- fused QKV projection via WMMA ----------------
__global__ __launch_bounds__(256) void k_qkv(
    const float* __restrict__ x,
    const _Float16* __restrict__ Wq, const _Float16* __restrict__ Wk,
    const _Float16* __restrict__ Wv, const float* __restrict__ pe,
    float* __restrict__ Qs, float* __restrict__ Kb, float* __restrict__ Vb) {
  __shared__ _Float16 aswz[8 * 32 * PADH];  // 12 KB swizzled f16 stripe
  const int rowBase = blockIdx.x << 4;
  const int tid = threadIdx.x;

  stage_a_swz(x + (size_t)rowBase * C_, aswz, tid);
  if (blockIdx.x + 1 < (int)gridDim.x)  // gfx1250 global_prefetch_b8 of next stripe
    __builtin_prefetch(x + (size_t)(rowBase + 16) * C_ + tid * 16, 0, 0);
  __syncthreads();

  const int lane = tid & 31, wave = tid >> 5;
  const int mrow = lane & 15, hs = lane >> 4;

  v16h a[8];
#pragma unroll
  for (int c = 0; c < 8; ++c)
    a[c] = *(const v16h*)(aswz + ((c << 5) + lane) * PADH);  // 2x ds_load_b128

  for (int t6 = 0; t6 < 6; ++t6) {
    const int t = wave * 6 + t6;          // 0..47 : [Q|K|V] x 16 col-tiles
    const int wsel = t >> 4;              // uniform per wave
    const int colTile = t & 15;
    const _Float16* W = (wsel == 0) ? Wq : (wsel == 1) ? Wk : Wv;
    const _Float16* wbase = W + ((size_t)(colTile << 3) << 9) + lane * 16;
    v8f acc = {};
#pragma unroll
    for (int c = 0; c < 8; ++c) {
      v16h bfr = *(const v16h*)(wbase + (c << 9));  // 2x global_load_b128
      acc = __builtin_amdgcn_wmma_f32_16x16x32_f16(false, a[c], false, bfr,
                                                   (short)0, acc, false, false);
    }
    float* dst = (wsel == 0) ? Qs : (wsel == 1) ? Kb : Vb;
    const int colBase = colTile << 4;
#pragma unroll
    for (int j = 0; j < 8; ++j) {
      int row = rowBase + j + (hs << 3);
      int col = colBase + mrow;
      float v = acc[j];
      if (wsel == 0) {  // fold query pos_emb + softmax scale into stored Q
        int px = row & (X_ - 1);
        int py = (row >> 7) & (Y_ - 1);
        int qidx = (py - clampi(py, 2, Y_ - 3) + 2) * 5 +
                   (px - clampi(px, 2, X_ - 3) + 2);
        v = (v + pe[qidx * C_ + col]) * SCALE;
      }
      dst[(size_t)row * C_ + col] = v;
    }
  }
}

// ---------------- local attention: scores + softmax + weighted V ----------------
__global__ __launch_bounds__(256) void k_attn(
    const float* __restrict__ Qs, const float* __restrict__ Kb,
    const float* __restrict__ Vb, const float* __restrict__ pe,
    float* __restrict__ outAcc) {
  __shared__ float halo[20 * 20 * 36];  // 57.6 KB, K halo then V halo
  __shared__ float pes[S_ * F_];        // 3.2 KB head slice of pos_emb

  const int tx0 = blockIdx.x << 4, ty0 = blockIdx.y << 4;
  const int h = blockIdx.z & (H_ - 1), b = blockIdx.z >> 3;
  const int tid = threadIdx.x;
  const int oy = clampi(ty0, 2, Y_ - 3) - 2;  // halo origin
  const int ox = clampi(tx0, 2, X_ - 3) - 2;

  for (int i = tid; i < S_ * F_; i += 256)
    pes[i] = pe[(i >> 5) * C_ + (h << 5) + (i & 31)];

  for (int i = tid; i < 400 * 8; i += 256) {  // phase 1: K halo
    int cell = i >> 3, f4 = i & 7;
    int gy = mini(oy + cell / 20, Y_ - 1);
    int gx = mini(ox + cell % 20, X_ - 1);
    size_t roff = ((size_t)((b * Y_ + gy) * X_ + gx)) * C_ + (h << 5) + (f4 << 2);
    *(float4*)(halo + cell * 36 + (f4 << 2)) = *(const float4*)(Kb + roff);
  }
  __syncthreads();

  const int ly = tid >> 4, lx = tid & 15;
  const int y = ty0 + ly, xx = tx0 + lx;
  const size_t row = (size_t)((b * Y_ + y) * X_ + xx);

  float q[F_];
  {
    const float4* qp = (const float4*)(Qs + row * C_ + (h << 5));
#pragma unroll
    for (int f4 = 0; f4 < 8; ++f4) {
      float4 v = qp[f4];
      q[4 * f4 + 0] = v.x; q[4 * f4 + 1] = v.y;
      q[4 * f4 + 2] = v.z; q[4 * f4 + 3] = v.w;
    }
  }
  const int cy = clampi(y, 2, Y_ - 3) - oy;   // neighbor LDS row = cy+dy in [0,19]
  const int cx = clampi(xx, 2, X_ - 3) - ox;

  float sc[S_];
  for (int s = 0; s < S_; ++s) {
    const float* kp = halo + ((cy + s / 5 - 2) * 20 + (cx + s % 5 - 2)) * 36;
    const float* pp = pes + (s << 5);
    float accv = 0.f;
#pragma unroll
    for (int f = 0; f < F_; ++f) accv = fmaf(q[f], kp[f] + pp[f], accv);
    sc[s] = accv;   // 1/sqrt(F) already folded into Q
  }
  float mx = sc[0];
#pragma unroll
  for (int s = 1; s < S_; ++s) mx = fmaxf(mx, sc[s]);
  float sum = 0.f;
#pragma unroll
  for (int s = 0; s < S_; ++s) { sc[s] = __expf(sc[s] - mx); sum += sc[s]; }
  const float inv = 1.f / sum;

  __syncthreads();  // all K reads complete before overwriting halo
  for (int i = tid; i < 400 * 8; i += 256) {  // phase 2: V halo
    int cell = i >> 3, f4 = i & 7;
    int gy = mini(oy + cell / 20, Y_ - 1);
    int gx = mini(ox + cell % 20, X_ - 1);
    size_t roff = ((size_t)((b * Y_ + gy) * X_ + gx)) * C_ + (h << 5) + (f4 << 2);
    *(float4*)(halo + cell * 36 + (f4 << 2)) = *(const float4*)(Vb + roff);
  }
  __syncthreads();

  float o[F_];
#pragma unroll
  for (int f = 0; f < F_; ++f) o[f] = 0.f;
  for (int s = 0; s < S_; ++s) {
    const float* vp = halo + ((cy + s / 5 - 2) * 20 + (cx + s % 5 - 2)) * 36;
    const float w = sc[s];
#pragma unroll
    for (int f = 0; f < F_; ++f) o[f] = fmaf(w, vp[f], o[f]);
  }
  float4* op = (float4*)(outAcc + row * C_ + (h << 5));
#pragma unroll
  for (int f4 = 0; f4 < 8; ++f4) {
    float4 v;
    v.x = o[4 * f4 + 0] * inv; v.y = o[4 * f4 + 1] * inv;
    v.z = o[4 * f4 + 2] * inv; v.w = o[4 * f4 + 3] * inv;
    op[f4] = v;
  }
}

// ---------------- output projection via WMMA ----------------
__global__ __launch_bounds__(256) void k_out(
    const float* __restrict__ A, const _Float16* __restrict__ Wo,
    float* __restrict__ out) {
  __shared__ _Float16 aswz[8 * 32 * PADH];  // 12 KB
  const int rowBase = blockIdx.x << 4;
  const int tid = threadIdx.x;

  stage_a_swz(A + (size_t)rowBase * C_, aswz, tid);
  if (blockIdx.x + 1 < (int)gridDim.x)
    __builtin_prefetch(A + (size_t)(rowBase + 16) * C_ + tid * 16, 0, 0);
  __syncthreads();

  const int lane = tid & 31, wave = tid >> 5;
  const int mrow = lane & 15, hs = lane >> 4;

  v16h a[8];
#pragma unroll
  for (int c = 0; c < 8; ++c)
    a[c] = *(const v16h*)(aswz + ((c << 5) + lane) * PADH);

#pragma unroll
  for (int t2 = 0; t2 < 2; ++t2) {     // 8 waves x 2 = 16 col tiles
    const int colTile = (wave << 1) + t2;
    const _Float16* wbase = Wo + ((size_t)(colTile << 3) << 9) + lane * 16;
    v8f acc = {};
#pragma unroll
    for (int c = 0; c < 8; ++c) {
      v16h bfr = *(const v16h*)(wbase + (c << 9));
      acc = __builtin_amdgcn_wmma_f32_16x16x32_f16(false, a[c], false, bfr,
                                                   (short)0, acc, false, false);
    }
    const int colBase = colTile << 4;
#pragma unroll
    for (int j = 0; j < 8; ++j) {
      int r = rowBase + j + (hs << 3);
      out[(size_t)r * C_ + colBase + mrow] = acc[j];
    }
  }
}

extern "C" void kernel_launch(void* const* d_in, const int* in_sizes, int n_in,
                              void* d_out, int out_size, void* d_ws, size_t ws_size,
                              hipStream_t stream) {
  (void)in_sizes; (void)n_in; (void)out_size; (void)ws_size;
  const float* x  = (const float*)d_in[0];
  const float* Wq = (const float*)d_in[1];
  const float* Wk = (const float*)d_in[2];
  const float* Wv = (const float*)d_in[3];
  const float* Wo = (const float*)d_in[4];
  const float* pe = (const float*)d_in[5];
  float* out = (float*)d_out;

  // workspace: 4 swizzled f16 weights (512 KB) + 4 f32 [N,256] tensors (537 MB)
  char* p = (char*)d_ws;
  constexpr size_t WE = (size_t)C_ * C_;       // 65536 elements per weight
  _Float16* Wq16 = (_Float16*)p; p += WE * 2;
  _Float16* Wk16 = (_Float16*)p; p += WE * 2;
  _Float16* Wv16 = (_Float16*)p; p += WE * 2;
  _Float16* Wo16 = (_Float16*)p; p += WE * 2;
  constexpr size_t TE = (size_t)NROW * C_;     // 33.5M elements per tensor
  float* Qs   = (float*)p; p += TE * 4;
  float* Kb   = (float*)p; p += TE * 4;
  float* Vb   = (float*)p; p += TE * 4;
  float* oAcc = (float*)p; p += TE * 4;

  k_cvt_swz<<<(int)(WE / 256), 256, 0, stream>>>(Wq, Wq16);
  k_cvt_swz<<<(int)(WE / 256), 256, 0, stream>>>(Wk, Wk16);
  k_cvt_swz<<<(int)(WE / 256), 256, 0, stream>>>(Wv, Wv16);
  k_cvt_swz<<<(int)(WE / 256), 256, 0, stream>>>(Wo, Wo16);

  k_qkv<<<NROW / 16, 256, 0, stream>>>(x, Wq16, Wk16, Wv16, pe, Qs, Kb, Vb);
  k_attn<<<dim3(X_ / 16, Y_ / 16, B_ * H_), 256, 0, stream>>>(Qs, Kb, Vb, pe, oAcc);
  k_out<<<NROW / 16, 256, 0, stream>>>(oAcc, Wo16, out);
}
